// ClassifierGridSearch_72859825209456
// MI455X (gfx1250) — matrix operand order repaired and checked by reading, not compiled, not run on hardware
//
#include <hip/hip_runtime.h>

#define NN 40000
#define NE 640000
#define NR 16
#define DD 128
#define DFC 256
#define NC 3
#define NG 64

typedef __attribute__((ext_vector_type(16))) __bf16   v16bf;
typedef __attribute__((ext_vector_type(8)))  float    v8f;
typedef __attribute__((ext_vector_type(4)))  unsigned u32x4;

union V16 { struct { u32x4 lo, hi; } u; v16bf v; };

__device__ __forceinline__ unsigned short f2bf(float f) {
  unsigned u = __float_as_uint(f);
  u += 0x7FFFu + ((u >> 16) & 1u);          // round-to-nearest-even
  return (unsigned short)(u >> 16);
}
__device__ __forceinline__ float bf2f(unsigned short h) {
  return __uint_as_float(((unsigned)h) << 16);
}
__device__ __forceinline__ void atomAddF(float* p, float v) {
  (void)__hip_atomic_fetch_add(p, v, __ATOMIC_RELAXED, __HIP_MEMORY_SCOPE_AGENT);
}
__device__ __forceinline__ v8f wmma_bf16(v16bf a, v16bf b, v8f c) {
  return __builtin_amdgcn_wmma_f32_16x16x32_bf16(false, a, false, b, (short)0, c,
                                                 false, false);
}
// A operand (16x32 bf16, MxK): lane m%16 holds row m; lanes>=16 carry K+8;
// VGPRs 0-3 = K koff..koff+7, VGPRs 4-7 = K koff+16..koff+23  -> two b128 loads.
__device__ __forceinline__ v16bf load_a(const unsigned short* hb, int row, int kb,
                                        int lane) {
  int koff = (lane & 16) ? 8 : 0;
  const unsigned short* p = hb + (size_t)row * DD + kb + koff;
  V16 t;
  t.u.lo = *(const u32x4*)p;
  t.u.hi = *(const u32x4*)(p + 16);
  return t.v;
}
// B operand pre-packed: 16 contiguous ushorts per lane per (kblock,ntile) tile.
// Works for both global and LDS source pointers.
__device__ __forceinline__ v16bf load_b(const unsigned short* pk, int kbi, int nt,
                                        int lane) {
  const unsigned short* p = pk + (((kbi * 8 + nt) * 32 + lane) << 4);
  V16 t;
  t.u.lo = *(const u32x4*)p;
  t.u.hi = *(const u32x4*)(p + 8);
  return t.v;
}

__global__ void k_init(int* zr, int n) {
  int i = blockIdx.x * 256 + threadIdx.x;
  if (i < n) zr[i] = 0;
}
__global__ void k_count(const int* rel, int* cnt) {
  int e = blockIdx.x * 256 + threadIdx.x;
  if (e < NE) atomicAdd(&cnt[rel[e]], 1);
}
__global__ void k_scan(const int* cnt, int* off, int* cur) {
  if (threadIdx.x == 0) {
    int o = 0;
    for (int r = 0; r < NR; ++r) {
      off[r] = o; cur[r] = o;
      o += cnt[r];
    }
    off[NR] = o;
  }
}
__global__ void k_bucket(const int* src, const int* dst, const int* rel, int* cur,
                         int* ssrc, int* sdst) {
  int e = blockIdx.x * 256 + threadIdx.x;
  if (e < NE) {
    int p = atomicAdd(&cur[rel[e]], 1);
    ssrc[p] = src[e];
    sdst[p] = dst[e];
  }
}
// Pack W[r][k][n] (f32, row-major 128x128) into WMMA B-operand bf16 layout:
// VGPR v holds K = kb + 2v(,+1) (+8 for v>=4), lanes 16-31 hold K+8, column n = nt*16+lane%16.
__global__ void k_pack(const float* W, unsigned* out, int R) {
  int tid = blockIdx.x * 256 + threadIdx.x;
  if (tid >= R * 8192) return;
  int v = tid & 7, lane = (tid >> 3) & 31, nt = (tid >> 8) & 7,
      kbi = (tid >> 11) & 3, r = tid >> 13;
  int k = kbi * 32 + 2 * v + ((lane & 16) ? 8 : 0) + ((v & 4) ? 8 : 0);
  int n = nt * 16 + (lane & 15);
  unsigned lo = f2bf(W[(size_t)(r * DD + k) * DD + n]);
  unsigned hi = f2bf(W[(size_t)(r * DD + k + 1) * DD + n]);
  out[tid] = lo | (hi << 16);
}
__global__ void k_f2bf(const float* x, unsigned short* y, int n) {
  int i = blockIdx.x * 256 + threadIdx.x;
  if (i < n) y[i] = f2bf(x[i]);
}
__global__ void k_relu_bf(const float* x, unsigned short* y, int n) {
  int i = blockIdx.x * 256 + threadIdx.x;
  if (i < n) y[i] = f2bf(fmaxf(x[i], 0.f));
}
// agg[16-row tile] = hb @ Ws + bias   (one 16x16 tile per wave, 8 waves = 128 cols)
__global__ void k_selfloop(const unsigned short* hb, const unsigned short* pk,
                           const float* bias, float* agg) {
  int lane = threadIdx.x & 31, nt = threadIdx.x >> 5;
  int mbase = blockIdx.x * 16;
  v8f c = {};
  #pragma unroll
  for (int kbi = 0; kbi < 4; ++kbi) {
    v16bf a = load_a(hb, mbase + (lane & 15), kbi * 32, lane);
    v16bf b = load_b(pk, kbi, nt, lane);
    c = wmma_bf16(a, b, c);
  }
  int n = nt * 16 + (lane & 15);
  float bn = bias[n];
  int mo = (lane & 16) ? 8 : 0;
  #pragma unroll
  for (int v = 0; v < 8; ++v)
    agg[(size_t)(mbase + mo + v) * DD + n] = c[v] + bn;
}
// Block = one relation (blockIdx.y) x one group of 8 edge-tiles (blockIdx.x).
// W_r staged once into LDS (32 KB); each wave does 16 edges: gathered A @ W_r,
// then atomic scatter-add of the 16x128 message tile into agg[dst].
__global__ void k_edge(const unsigned short* hb, const unsigned short* pkW,
                       const int* ssrc, const int* sdst, const int* roff,
                       float* agg) {
  __shared__ __align__(16) unsigned short bsh[16384];   // 32 KB packed W_r
  int r = blockIdx.y;
  int base = roff[r];
  int cnt = roff[r + 1] - base;
  int tiles = (cnt + 15) >> 4;
  int tile0 = blockIdx.x * 8;
  if (tile0 >= tiles) return;               // block-uniform early exit
  {
    const u32x4* s4 = (const u32x4*)(pkW + (size_t)r * 16384);
    u32x4* d4 = (u32x4*)bsh;
    #pragma unroll
    for (int i = 0; i < 8; ++i)             // 2048 u32x4 / 256 threads
      d4[threadIdx.x + 256 * i] = s4[threadIdx.x + 256 * i];
  }
  __syncthreads();
  int lane = threadIdx.x & 31, wave = threadIdx.x >> 5;
  int t = tile0 + wave;
  if (t >= tiles) return;                   // no barriers after this point
  int ebase = base + t * 16;
  int ecnt = cnt - t * 16;
  if (ecnt > 16) ecnt = 16;
  int m16 = lane & 15;
  int srow = ssrc[ebase + (m16 < ecnt ? m16 : 0)];
  v8f c[8] = {};
  for (int kbi = 0; kbi < 4; ++kbi) {
    v16bf a = load_a(hb, srow, kbi * 32, lane);
    #pragma unroll
    for (int nt = 0; nt < 8; ++nt) {
      v16bf b = load_b(bsh, kbi, nt, lane); // ds_load_b128 x2
      c[nt] = wmma_bf16(a, b, c[nt]);
    }
  }
  int mo = (lane & 16) ? 8 : 0;
  #pragma unroll
  for (int v = 0; v < 8; ++v) {
    int m = mo + v;
    if (m < ecnt) {
      size_t drow = (size_t)sdst[ebase + m] * DD;
      #pragma unroll
      for (int nt = 0; nt < 8; ++nt)
        atomAddF(&agg[drow + nt * 16 + m16], c[nt][v]);
    }
  }
}
__global__ void k_pool(const unsigned short* hb, const int* gid, float* psum,
                       int* pcnt) {
  int i = blockIdx.x * 256 + threadIdx.x;
  if (i < NN * DD) {
    int nrow = i >> 7, d = i & 127;
    int g = gid[nrow];
    atomAddF(&psum[g * DD + d], bf2f(hb[i]));
    if (d == 0) atomicAdd(&pcnt[g], 1);
  }
}
__global__ void k_head(const float* psum, const int* pcnt, const float* Wfc,
                       const float* bfc, const float* Wc, const float* bc,
                       float* out) {
  __shared__ float hg[DD];
  __shared__ float sfc[DFC];
  int g = blockIdx.x, t = threadIdx.x;
  float inv = 1.f / fmaxf((float)pcnt[g], 1.f);
  if (t < DD) hg[t] = psum[g * DD + t] * inv;
  __syncthreads();
  float acc = bfc[t];
  for (int d = 0; d < DD; ++d) acc += hg[d] * Wfc[d * DFC + t];
  sfc[t] = fmaxf(acc, 0.f);
  __syncthreads();
  if (t == 0) {
    float lg[NC];
    for (int k = 0; k < NC; ++k) {
      float a = bc[k];
      for (int j = 0; j < DFC; ++j) a += sfc[j] * Wc[j * NC + k];
      lg[k] = a;
    }
    float mx = fmaxf(lg[0], fmaxf(lg[1], lg[2]));
    float s = 0.f, ex[NC];
    for (int k = 0; k < NC; ++k) { ex[k] = expf(lg[k] - mx); s += ex[k]; }
    for (int k = 0; k < NC; ++k) out[g * NC + k] = ex[k] / s;
  }
}

extern "C" void kernel_launch(void* const* d_in, const int* in_sizes, int n_in,
                              void* d_out, int out_size, void* d_ws, size_t ws_size,
                              hipStream_t stream) {
  (void)in_sizes; (void)n_in; (void)out_size; (void)ws_size;
  const float* h   = (const float*)d_in[0];
  const int*   src = (const int*)d_in[1];
  const int*   dst = (const int*)d_in[2];
  const int*   rel = (const int*)d_in[3];
  const int*   gid = (const int*)d_in[4];
  const float* W1  = (const float*)d_in[5];
  const float* Ws1 = (const float*)d_in[6];
  const float* b1  = (const float*)d_in[7];
  const float* W2  = (const float*)d_in[8];
  const float* Ws2 = (const float*)d_in[9];
  const float* b2  = (const float*)d_in[10];
  const float* Wfc = (const float*)d_in[11];
  const float* bfc = (const float*)d_in[12];
  const float* Wc  = (const float*)d_in[13];
  const float* bc  = (const float*)d_in[14];
  float* out = (float*)d_out;

  char* ws   = (char*)d_ws;                       // ~37 MB used
  int*   meta = (int*)ws;
  float* psum = (float*)ws;                       // [8192] f32
  int*   pcnt = meta + 8192;                      // [64]
  int*   rcnt = meta + 8256;                      // [16]
  int*   rcur = meta + 8272;                      // [16]
  int*   roff = meta + 8288;                      // [17]
  int*   ssrc = (int*)(ws + 36864);               // [NE]
  int*   sdst = (int*)(ws + 2596864);             // [NE]
  unsigned short* hb   = (unsigned short*)(ws + 5156864);   // [NN*DD] bf16
  float*          agg  = (float*)(ws + 15396864);           // [NN*DD] f32
  unsigned short* pW1  = (unsigned short*)(ws + 35876864);  // 16 x 32KB
  unsigned short* pW2  = (unsigned short*)(ws + 36401152);
  unsigned short* pWs1 = (unsigned short*)(ws + 36925440);
  unsigned short* pWs2 = (unsigned short*)(ws + 36958208);

  k_init<<<33, 256, 0, stream>>>(meta, 8288);
  k_count<<<(NE + 255) / 256, 256, 0, stream>>>(rel, rcnt);
  k_scan<<<1, 32, 0, stream>>>(rcnt, roff, rcur);
  k_bucket<<<(NE + 255) / 256, 256, 0, stream>>>(src, dst, rel, rcur, ssrc, sdst);
  k_pack<<<512, 256, 0, stream>>>(W1, (unsigned*)pW1, NR);
  k_pack<<<512, 256, 0, stream>>>(W2, (unsigned*)pW2, NR);
  k_pack<<<32, 256, 0, stream>>>(Ws1, (unsigned*)pWs1, 1);
  k_pack<<<32, 256, 0, stream>>>(Ws2, (unsigned*)pWs2, 1);
  k_f2bf<<<(NN * DD + 255) / 256, 256, 0, stream>>>(h, hb, NN * DD);

  // grid.x covers worst case: one relation owning all edges -> 40000 tiles / 8
  dim3 egrid(NE / 16 / 8, NR);                    // (5000, 16)
  // layer 1
  k_selfloop<<<NN / 16, 256, 0, stream>>>(hb, pWs1, b1, agg);
  k_edge<<<egrid, 256, 0, stream>>>(hb, pW1, ssrc, sdst, roff, agg);
  k_relu_bf<<<(NN * DD + 255) / 256, 256, 0, stream>>>(agg, hb, NN * DD);
  // layer 2
  k_selfloop<<<NN / 16, 256, 0, stream>>>(hb, pWs2, b2, agg);
  k_edge<<<egrid, 256, 0, stream>>>(hb, pW2, ssrc, sdst, roff, agg);
  k_relu_bf<<<(NN * DD + 255) / 256, 256, 0, stream>>>(agg, hb, NN * DD);
  // mean-pool + FC + softmax
  k_pool<<<(NN * DD + 255) / 256, 256, 0, stream>>>(hb, gid, psum, pcnt);
  k_head<<<NG, 256, 0, stream>>>(psum, pcnt, Wfc, bfc, Wc, bc, out);
}
